// MultiHeadSelfAttention_46909632807067
// MI455X (gfx1250) — matrix operand order repaired and checked by reading, not compiled, not run on hardware
//
#include <hip/hip_runtime.h>
#include <hip/hip_bf16.h>

#define B_  2
#define S_  4096
#define D_  768
#define H_  12
#define DH_ 64

typedef __bf16 bf16_t;
typedef __attribute__((ext_vector_type(16))) __bf16 v16bf;
typedef __attribute__((ext_vector_type(8)))  __bf16 v8bf;
typedef __attribute__((ext_vector_type(4)))  __bf16 v4bf;
typedef __attribute__((ext_vector_type(8)))  float  v8f;

union AB16 { v16bf v; v8bf h[2]; };

// A operand (16x32 bf16, M x K), source row-major. Lane ln=lane&15 selects row.
// vector elems 0..7  = cols colChunk+(hi?8:0)    .. +7
// vector elems 8..15 = cols colChunk+(hi?8:0)+16 .. +23
static __device__ inline v16bf load_matA(const bf16_t* rowBase, int colChunk, int hi) {
    const bf16_t* p = rowBase + colChunk + (hi ? 8 : 0);
    AB16 u;
    u.h[0] = *(const v8bf*)(p);
    u.h[1] = *(const v8bf*)(p + 16);
    return u.v;
}

// B operand (32x16 bf16, K x N), stored N-major ([N][K] row-major); rowBase = row n.
// vector elems 0..7  = kdim colChunk+(hi?8:0)    .. +7
// vector elems 8..15 = kdim colChunk+(hi?8:0)+16 .. +23
static __device__ inline v16bf load_matB(const bf16_t* rowBase, int colChunk, int hi) {
    const bf16_t* p = rowBase + colChunk + (hi ? 8 : 0);
    AB16 u;
    u.h[0] = *(const v8bf*)(p);
    u.h[1] = *(const v8bf*)(p + 16);
    return u.v;
}

static __device__ inline v8f wmma_bf16(v16bf a, v16bf b, v8f c) {
    return __builtin_amdgcn_wmma_f32_16x16x32_bf16(false, a, false, b, (short)0, c, false, false);
}

// raw v_exp_f32 (base-2); inputs are always <= 0 here, deep-negative flushes to 0.
static __device__ inline float fast_exp2(float x) { return __builtin_amdgcn_exp2f(x); }

// ---------------- conversion kernels ----------------

__global__ __launch_bounds__(256) void convert_x_kernel(const float* __restrict__ x,
                                                        bf16_t* __restrict__ xb, int n4) {
    int i = blockIdx.x * blockDim.x + threadIdx.x;
    if (i >= n4) return;
    const float4 f = ((const float4*)x)[i];
    v4bf o;
    o[0] = (bf16_t)f.x; o[1] = (bf16_t)f.y; o[2] = (bf16_t)f.z; o[3] = (bf16_t)f.w;
    *(v4bf*)(xb + (size_t)i * 4) = o;
}

// Wt layout: [3][H][DH][D] bf16, transposed from [H][D][DH].
// Wq additionally carries 1/sqrt(DH) * log2(e): scores then come out of the
// QK^T WMMA directly in the base-2 domain, so softmax uses raw v_exp_f32.
__global__ __launch_bounds__(256) void convert_w_kernel(const float* __restrict__ Wq,
                                                        const float* __restrict__ Wk,
                                                        const float* __restrict__ Wv,
                                                        bf16_t* __restrict__ Wt) {
    const int per = H_ * DH_ * D_;
    int tid = blockIdx.x * blockDim.x + threadIdx.x;
    if (tid >= 3 * per) return;
    int m = tid / per;
    int o = tid % per;
    int h = o / (DH_ * D_);
    int r = o % (DH_ * D_);
    int e = r / D_;
    int d = r % D_;
    const float* src = (m == 0) ? Wq : (m == 1) ? Wk : Wv;
    float v = src[((size_t)h * D_ + d) * DH_ + e];
    if (m == 0) v *= 0.125f * 1.44269504088896340736f;  // 1/sqrt(64) * log2(e)
    Wt[tid] = (bf16_t)v;
}

// ---------------- QKV projection GEMM ----------------
// grid: B*H*(S/64), block: 128 (4 waves). Wave w: 16 s-rows x DH=64 for Q,K,V.
// Copy-free double buffering: manual unroll-by-2 over the 24 D-chunks with
// explicit even/odd buffer sets (no register rotation -> no WAR-hazard NOPs).
__global__ __launch_bounds__(128) void qkv_gemm_kernel(const bf16_t* __restrict__ xb,
                                                       const bf16_t* __restrict__ Wt,
                                                       bf16_t* __restrict__ Q,
                                                       bf16_t* __restrict__ K,
                                                       bf16_t* __restrict__ Vt) {
    const int nSB = S_ / 64;
    int blk  = blockIdx.x;
    int sb   = blk % nSB;
    int bh   = blk / nSB;          // b*H + h
    int h    = bh % H_;
    int b    = bh / H_;
    int wave = threadIdx.x >> 5;
    int lane = threadIdx.x & 31;
    int hi   = lane >> 4;
    int ln   = lane & 15;

    v8f accQ[4] = {}, accK[4] = {}, accV[4] = {};

    const int srowA = sb * 64 + wave * 16 + ln;
    const bf16_t* xrow = xb + ((size_t)b * S_ + srowA) * D_;
    const size_t per = (size_t)H_ * DH_ * D_;
    const bf16_t* wq = Wt + (size_t)h * DH_ * D_;
    const bf16_t* wk = wq + per;
    const bf16_t* wv = wk + per;

    size_t nrow[4];
    #pragma unroll
    for (int t = 0; t < 4; ++t) nrow[t] = (size_t)(16 * t + ln) * D_;

    v16bf a0, a1;
    v16bf b0q[4], b0k[4], b0v[4];
    v16bf b1q[4], b1k[4], b1v[4];

    // prologue: chunk 0 into even buffers
    a0 = load_matA(xrow, 0, hi);
    #pragma unroll
    for (int t = 0; t < 4; ++t) {
        b0q[t] = load_matB(wq + nrow[t], 0, hi);
        b0k[t] = load_matB(wk + nrow[t], 0, hi);
        b0v[t] = load_matB(wv + nrow[t], 0, hi);
    }

    // pairs: iteration i computes chunks 2i (even buf) and 2i+1 (odd buf),
    // while loading 2i+1 (odd) and 2i+2 (even).
    for (int i = 0; i < D_ / 64 - 1; ++i) {   // 11 iterations
        const int cOdd  = i * 64 + 32;
        const int cEven = i * 64 + 64;

        a1 = load_matA(xrow, cOdd, hi);
        #pragma unroll
        for (int t = 0; t < 4; ++t) {
            b1q[t] = load_matB(wq + nrow[t], cOdd, hi);
            b1k[t] = load_matB(wk + nrow[t], cOdd, hi);
            b1v[t] = load_matB(wv + nrow[t], cOdd, hi);
        }
        #pragma unroll
        for (int t = 0; t < 4; ++t) {
            accQ[t] = wmma_bf16(a0, b0q[t], accQ[t]);
            accK[t] = wmma_bf16(a0, b0k[t], accK[t]);
            accV[t] = wmma_bf16(a0, b0v[t], accV[t]);
        }

        a0 = load_matA(xrow, cEven, hi);
        #pragma unroll
        for (int t = 0; t < 4; ++t) {
            b0q[t] = load_matB(wq + nrow[t], cEven, hi);
            b0k[t] = load_matB(wk + nrow[t], cEven, hi);
            b0v[t] = load_matB(wv + nrow[t], cEven, hi);
        }
        #pragma unroll
        for (int t = 0; t < 4; ++t) {
            accQ[t] = wmma_bf16(a1, b1q[t], accQ[t]);
            accK[t] = wmma_bf16(a1, b1k[t], accK[t]);
            accV[t] = wmma_bf16(a1, b1v[t], accV[t]);
        }
    }

    // epilogue: chunks 22 (even buf, already loaded) and 23 (odd buf)
    {
        const int cOdd = D_ - 32;   // 736
        a1 = load_matA(xrow, cOdd, hi);
        #pragma unroll
        for (int t = 0; t < 4; ++t) {
            b1q[t] = load_matB(wq + nrow[t], cOdd, hi);
            b1k[t] = load_matB(wk + nrow[t], cOdd, hi);
            b1v[t] = load_matB(wv + nrow[t], cOdd, hi);
        }
        #pragma unroll
        for (int t = 0; t < 4; ++t) {
            accQ[t] = wmma_bf16(a0, b0q[t], accQ[t]);
            accK[t] = wmma_bf16(a0, b0k[t], accK[t]);
            accV[t] = wmma_bf16(a0, b0v[t], accV[t]);
        }
        #pragma unroll
        for (int t = 0; t < 4; ++t) {
            accQ[t] = wmma_bf16(a1, b1q[t], accQ[t]);
            accK[t] = wmma_bf16(a1, b1k[t], accK[t]);
            accV[t] = wmma_bf16(a1, b1v[t], accV[t]);
        }
    }

    const int srow0 = sb * 64 + wave * 16 + hi * 8;  // C-layout: row = r + 8*hi
    bf16_t* Qb = Q  + (size_t)bh * S_ * DH_;
    bf16_t* Kb = K  + (size_t)bh * S_ * DH_;
    bf16_t* Vb = Vt + (size_t)bh * DH_ * S_;
    #pragma unroll
    for (int t = 0; t < 4; ++t) {
        int dh = 16 * t + ln;
        #pragma unroll
        for (int r = 0; r < 8; ++r) {
            int s = srow0 + r;
            Qb[(size_t)s * DH_ + dh] = (bf16_t)accQ[t][r];
            Kb[(size_t)s * DH_ + dh] = (bf16_t)accK[t][r];
            Vb[(size_t)dh * S_ + s]  = (bf16_t)accV[t][r];   // V stored transposed
        }
    }
}

// ---------------- flash attention ----------------
// grid: B*H*(S/64), block: 128 (4 waves). Wave w owns q rows [qb*64+16w, +16).
// Scores arrive pre-scaled by log2(e)/sqrt(DH): softmax runs in base-2.
__global__ __launch_bounds__(128) void attn_kernel(const bf16_t* __restrict__ Q,
                                                   const bf16_t* __restrict__ K,
                                                   const bf16_t* __restrict__ Vt,
                                                   float* __restrict__ out) {
    const int nQB = S_ / 64;
    int blk  = blockIdx.x;
    int qb   = nQB - 1 - (blk % nQB);   // heaviest q-blocks first (causal load balance)
    int bh   = blk / nQB;
    int h    = bh % H_;
    int b    = bh / H_;
    int wave = threadIdx.x >> 5;
    int lane = threadIdx.x & 31;
    int hi   = lane >> 4;
    int ln   = lane & 15;

    __shared__ __align__(16) bf16_t Pb[4][16][64];   // wave-private P tiles (8 KB)

    const bf16_t* Qh = Q  + (size_t)bh * S_ * DH_;
    const bf16_t* Kh = K  + (size_t)bh * S_ * DH_;
    const bf16_t* Vh = Vt + (size_t)bh * DH_ * S_;

    const int qrowA = qb * 64 + wave * 16 + ln;
    v16bf qa0 = load_matA(Qh + (size_t)qrowA * DH_, 0,  hi);
    v16bf qa1 = load_matA(Qh + (size_t)qrowA * DH_, 32, hi);

    v8f acc[4] = {};
    float mrow[8], lrow[8];
    #pragma unroll
    for (int r = 0; r < 8; ++r) { mrow[r] = -1e30f; lrow[r] = 0.0f; }

    const int q0 = qb * 64 + wave * 16 + hi * 8;  // first of the 8 rows this lane holds

    for (int kb = 0; kb <= qb; ++kb) {
        const int kbase = kb * 64;

        // batch-load all K and V operands for this key block into distinct registers:
        // V loads are independent of the score math, so they overlap softmax latency.
        v16bf kB[4][2], vB[4][2];
        #pragma unroll
        for (int t = 0; t < 4; ++t) {
            const bf16_t* krow = Kh + (size_t)(kbase + 16 * t + ln) * DH_;
            kB[t][0] = load_matB(krow, 0,  hi);
            kB[t][1] = load_matB(krow, 32, hi);
            const bf16_t* vrow = Vh + (size_t)(16 * t + ln) * S_ + kbase;
            vB[t][0] = load_matB(vrow, 0,  hi);
            vB[t][1] = load_matB(vrow, 32, hi);
        }

        // prefetch next key block (K rows + V^T row segments) into cache
        if (kb < qb) {
            const int nb = kbase + 64;
            #pragma unroll
            for (int t = 0; t < 4; ++t) {
                __builtin_prefetch(Kh + (size_t)(nb + 16 * t + ln) * DH_, 0, 0);
                __builtin_prefetch(Vh + (size_t)(16 * t + ln) * S_ + nb, 0, 0);
            }
        }

        // scores: 16q x 64k as 4 tiles, K-dim = DH in two k=32 chunks
        v8f sc[4];
        #pragma unroll
        for (int t = 0; t < 4; ++t) {
            v8f z = {};
            z = wmma_bf16(qa0, kB[t][0], z);
            z = wmma_bf16(qa1, kB[t][1], z);
            sc[t] = z;
        }

        // causal mask on the diagonal block
        if (kb == qb) {
            #pragma unroll
            for (int t = 0; t < 4; ++t) {
                int key = kbase + 16 * t + ln;
                #pragma unroll
                for (int r = 0; r < 8; ++r)
                    if (key > q0 + r) sc[t][r] = -1e30f;
            }
        }

        // online softmax (base-2): row max via 16-lane xor-shuffle reduction
        #pragma unroll
        for (int r = 0; r < 8; ++r) {
            float v = fmaxf(fmaxf(sc[0][r], sc[1][r]), fmaxf(sc[2][r], sc[3][r]));
            v = fmaxf(v, __shfl_xor(v, 1, 32));
            v = fmaxf(v, __shfl_xor(v, 2, 32));
            v = fmaxf(v, __shfl_xor(v, 4, 32));
            v = fmaxf(v, __shfl_xor(v, 8, 32));
            float mn    = fmaxf(mrow[r], v);
            float alpha = fast_exp2(mrow[r] - mn);     // single v_exp_f32
            mrow[r] = mn;
            lrow[r] *= alpha;
            #pragma unroll
            for (int t = 0; t < 4; ++t) acc[t][r] *= alpha;
        }

        // p = 2^(s - m), row sum
        #pragma unroll
        for (int r = 0; r < 8; ++r) {
            float rs = 0.0f;
            #pragma unroll
            for (int t = 0; t < 4; ++t) {
                float p = fast_exp2(sc[t][r] - mrow[r]);   // single v_exp_f32
                sc[t][r] = p;
                rs += p;
            }
            rs += __shfl_xor(rs, 1, 32);
            rs += __shfl_xor(rs, 2, 32);
            rs += __shfl_xor(rs, 4, 32);
            rs += __shfl_xor(rs, 8, 32);
            lrow[r] += rs;
        }

        // P (C-layout f32) -> LDS bf16 row-major [16][64]; per-wave region, DS in-order
        #pragma unroll
        for (int t = 0; t < 4; ++t) {
            #pragma unroll
            for (int r = 0; r < 8; ++r)
                Pb[wave][hi * 8 + r][16 * t + ln] = (bf16_t)sc[t][r];
        }
        v16bf pa0 = load_matA(&Pb[wave][ln][0], 0,  hi);
        v16bf pa1 = load_matA(&Pb[wave][ln][0], 32, hi);

        // ctx += P * V ; V^T rows give B operand directly (already resident in regs)
        #pragma unroll
        for (int t = 0; t < 4; ++t) {
            acc[t] = wmma_bf16(pa0, vB[t][0], acc[t]);
            acc[t] = wmma_bf16(pa1, vB[t][1], acc[t]);
        }
    }

    // finalize: divide by row sums (raw v_rcp_f32), write f32 output [B][S][H*DH]
    #pragma unroll
    for (int r = 0; r < 8; ++r) {
        float inv = __builtin_amdgcn_rcpf(lrow[r]);
        int s = q0 + r;
        float* op = out + ((size_t)b * S_ + s) * D_ + h * DH_;
        #pragma unroll
        for (int t = 0; t < 4; ++t)
            op[16 * t + ln] = acc[t][r] * inv;
    }
}

// ---------------- launcher ----------------

extern "C" void kernel_launch(void* const* d_in, const int* in_sizes, int n_in,
                              void* d_out, int out_size, void* d_ws, size_t ws_size,
                              hipStream_t stream) {
    (void)in_sizes; (void)n_in; (void)out_size; (void)ws_size;
    const float* x  = (const float*)d_in[0];
    const float* Wq = (const float*)d_in[1];
    const float* Wk = (const float*)d_in[2];
    const float* Wv = (const float*)d_in[3];
    float* out = (float*)d_out;

    char* ws = (char*)d_ws;
    size_t off = 0;
    auto wsalloc = [&](size_t bytes) -> char* {
        char* p = ws + off;
        off = (off + bytes + 255) & ~(size_t)255;
        return p;
    };
    bf16_t* xb  = (bf16_t*)wsalloc((size_t)B_ * S_ * D_ * 2);           // 12.6 MB
    bf16_t* Wt  = (bf16_t*)wsalloc((size_t)3 * H_ * DH_ * D_ * 2);      //  3.5 MB
    bf16_t* Qd  = (bf16_t*)wsalloc((size_t)B_ * H_ * S_ * DH_ * 2);     // 12.6 MB
    bf16_t* Kd  = (bf16_t*)wsalloc((size_t)B_ * H_ * S_ * DH_ * 2);     // 12.6 MB
    bf16_t* Vtd = (bf16_t*)wsalloc((size_t)B_ * H_ * S_ * DH_ * 2);     // 12.6 MB

    const int n4 = B_ * S_ * D_ / 4;
    convert_x_kernel<<<(n4 + 255) / 256, 256, 0, stream>>>(x, xb, n4);

    const int nw = 3 * H_ * DH_ * D_;
    convert_w_kernel<<<(nw + 255) / 256, 256, 0, stream>>>(Wq, Wk, Wv, Wt);

    const int nblk = B_ * H_ * (S_ / 64);
    qkv_gemm_kernel<<<nblk, 128, 0, stream>>>(xb, Wt, Qd, Kd, Vtd);
    attn_kernel<<<nblk, 128, 0, stream>>>(Qd, Kd, Vtd, out);
}
